// MambaBackbone_87359634801438
// MI455X (gfx1250) — matrix operand order repaired and checked by reading, not compiled, not run on hardware
//
#include <hip/hip_runtime.h>
#include <hip/hip_bf16.h>
#include <math.h>

// ---------------------------------------------------------------------------
// Problem constants (from the reference)
// ---------------------------------------------------------------------------
constexpr int DMODEL  = 1024;
constexpr int NLAYERS = 4;
constexpr int DINNER  = 2 * DMODEL;   // 2048
constexpr int DSTATE  = 16;
constexpr int DCONV   = 4;
constexpr int DTRANK  = DMODEL / 16;  // 64
constexpr int BB      = 2;
constexpr int TT      = 1024;
constexpr int ROWS    = BB * TT;      // 2048

// ---------------------------------------------------------------------------
// WMMA vector types (wave32, CDNA5)
// ---------------------------------------------------------------------------
typedef __attribute__((ext_vector_type(16))) __bf16 bf16x16;
typedef __attribute__((ext_vector_type(8)))  __bf16 bf16x8;
typedef __attribute__((ext_vector_type(8)))  float  f32x8;

__device__ __forceinline__ __bf16 f2bf(float f) {
  union { float f; unsigned u; } v; v.f = f;
  unsigned r = v.u + 0x7FFFu + ((v.u >> 16) & 1u);   // round-to-nearest-even
  unsigned short h = (unsigned short)(r >> 16);
  __bf16 o; __builtin_memcpy(&o, &h, 2); return o;
}

__device__ __forceinline__ float sigmoidf_(float x) {
  return 1.0f / (1.0f + __expf(-x));
}

// A-fragment: 16-bit A 16x32 layout — per lane, elements 0..7 hold K=off..off+7,
// elements 8..15 hold K=off+16..off+23 (off = 0 for lanes 0-15, 8 for lanes 16-31).
__device__ __forceinline__ bf16x16 load_a_frag(const __bf16* p) {
  bf16x8 lo = *reinterpret_cast<const bf16x8*>(p);
  bf16x8 hi = *reinterpret_cast<const bf16x8*>(p + 16);
  bf16x16 r;
#pragma unroll
  for (int i = 0; i < 8; ++i) { r[i] = lo[i]; r[i + 8] = hi[i]; }
  return r;
}

// ---------------------------------------------------------------------------
// bf16 WMMA GEMM:  C[M,N] (f32, row-major) = A[M,K] (bf16 rm) * W[N,K]^T (bf16 rm)
// One wave computes a (MT*16) x (NT*16) register tile: MT*NT wmma per K-step.
// N and K are compile-time so fragment/store addresses fold into the 24-bit
// instruction offsets (2 base address pairs total). __launch_bounds__(256,1)
// raises the VGPR budget so the 2x4 tile stays in registers (no spills).
// 8 waves / block tile the M dimension. ACC=true fuses "C += result".
// ---------------------------------------------------------------------------
template <int MT, int NT, int N, int K, bool ACC>
__global__ __launch_bounds__(256, 1)
void wmma_gemm_bf16(const __bf16* __restrict__ A,
                    const __bf16* __restrict__ W,
                    float* __restrict__ C) {
  const int lane = threadIdx.x & 31;
  const int wave = threadIdx.x >> 5;
  const int m0 = (blockIdx.y * 8 + wave) * (MT * 16);
  const int n0 = blockIdx.x * (NT * 16);

  const int half = lane >> 4;                // 0: lanes 0-15, 1: lanes 16-31
  const int lrow = lane & 15;
  const int aoff = half ? 8 : 0;             // A interleave offset
  const int boff = half ? 8 : 0;             // B interleave offset (same as A)

  const __bf16* Abase = A + (size_t)(m0 + lrow) * K + aoff;
  const __bf16* Wbase = W + (size_t)(n0 + lrow) * K + boff;

  f32x8 acc[MT][NT] = {};

  for (int k0 = 0; k0 < K; k0 += 32) {
    // speculative prefetch of a future K-chunk (global_prefetch_b8)
    __builtin_prefetch(Abase + k0 + 256, 0, 1);
    __builtin_prefetch(Wbase + k0 + 256, 0, 1);

    bf16x16 a[MT];
#pragma unroll
    for (int i = 0; i < MT; ++i)
      a[i] = load_a_frag(Abase + i * 16 * K + k0);   // constant tile offsets

    bf16x16 b[NT];
#pragma unroll
    for (int j = 0; j < NT; ++j)
      b[j] = load_a_frag(Wbase + j * 16 * K + k0);

#pragma unroll
    for (int i = 0; i < MT; ++i)
#pragma unroll
      for (int j = 0; j < NT; ++j)
        acc[i][j] = __builtin_amdgcn_wmma_f32_16x16x32_bf16(
            false, a[i], false, b[j], (short)0, acc[i][j], false, false);
  }

  // C/D layout: VGPR v, lane l -> m = tile_m + v + 8*half, n = tile_n + (l&15)
#pragma unroll
  for (int i = 0; i < MT; ++i) {
    float* Cbase = C + (size_t)(m0 + i * 16 + half * 8) * N + n0 + lrow;
#pragma unroll
    for (int v = 0; v < 8; ++v) {
#pragma unroll
      for (int j = 0; j < NT; ++j) {
        float* p = Cbase + v * N + j * 16;           // constant offsets
        if (ACC) *p += acc[i][j][v];
        else     *p  = acc[i][j][v];
      }
    }
  }
}

// ---------------------------------------------------------------------------
// Elementwise / conversion kernels
// ---------------------------------------------------------------------------
__global__ void cvt_f32_bf16_kernel(const float* __restrict__ src,
                                    __bf16* __restrict__ dst, int n) {
  int i = blockIdx.x * blockDim.x + threadIdx.x;
  if (i < n) dst[i] = f2bf(src[i]);
}

// depthwise causal conv (width 4) + bias + SiLU; emits f32 and bf16 copies
__global__ void conv_silu_kernel(const float* __restrict__ xz,   // (ROWS, 2*DINNER)
                                 const float* __restrict__ cw,   // (DINNER, 4)
                                 const float* __restrict__ cb,   // (DINNER)
                                 float* __restrict__ uc,         // (ROWS, DINNER)
                                 __bf16* __restrict__ uc_bf) {
  int idx = blockIdx.x * blockDim.x + threadIdx.x;
  if (idx >= ROWS * DINNER) return;
  int d = idx % DINNER;
  int t = (idx / DINNER) % TT;
  int b = idx / (DINNER * TT);
  const float* w = cw + d * DCONV;
  float acc = cb[d];
#pragma unroll
  for (int j = 0; j < DCONV; ++j) {
    int tt = t - (DCONV - 1) + j;
    if (tt >= 0)
      acc += w[j] * xz[(size_t)(b * TT + tt) * (2 * DINNER) + d];
  }
  float s = acc * sigmoidf_(acc);
  uc[idx]    = s;
  uc_bf[idx] = f2bf(s);
}

// extract dt_r = x_dbl[:, :DTRANK] as bf16 (strided gather + convert)
__global__ void dtr_extract_kernel(const float* __restrict__ x_dbl, // (ROWS,96)
                                   __bf16* __restrict__ dtr) {      // (ROWS,64)
  int idx = blockIdx.x * blockDim.x + threadIdx.x;
  if (idx >= ROWS * DTRANK) return;
  int r   = idx % DTRANK;
  int row = idx / DTRANK;
  dtr[idx] = f2bf(x_dbl[(size_t)row * (DTRANK + 2 * DSTATE) + r]);
}

// dt = softplus(dt_lin + dt_bias), in place
__global__ void softplus_kernel(float* __restrict__ dt,
                                const float* __restrict__ bias) {
  int idx = blockIdx.x * blockDim.x + threadIdx.x;
  if (idx >= ROWS * DINNER) return;
  int d = idx % DINNER;
  float v = dt[idx] + bias[d];
  dt[idx] = (v > 20.0f) ? v : log1pf(__expf(v));
}

// ---------------------------------------------------------------------------
// Selective scan: one thread per (batch, channel). h[16] + A_d[16] in regs.
// Shared per-timestep B/C vectors staged through LDS in chunks of 16 steps.
// ---------------------------------------------------------------------------
constexpr int TCHUNK = 16;

__global__ void scan_kernel(const float* __restrict__ x_dbl, // (ROWS, 96)
                            const float* __restrict__ dt,    // (ROWS, DINNER)
                            const float* __restrict__ uc,    // (ROWS, DINNER)
                            const float* __restrict__ A_log, // (DINNER, DSTATE)
                            float* __restrict__ ys) {        // (ROWS, DINNER)
  __shared__ float bc[TCHUNK][2 * DSTATE];
  const int d = blockIdx.x * blockDim.x + threadIdx.x;
  const int b = blockIdx.y;

  float h[DSTATE], Ad[DSTATE];
#pragma unroll
  for (int s = 0; s < DSTATE; ++s) {
    h[s]  = 0.0f;
    Ad[s] = -__expf(A_log[(size_t)d * DSTATE + s]);
  }

  for (int t0 = 0; t0 < TT; t0 += TCHUNK) {
    for (int i = threadIdx.x; i < TCHUNK * 2 * DSTATE; i += blockDim.x) {
      int j = i >> 5, c = i & 31;
      bc[j][c] = x_dbl[(size_t)(b * TT + t0 + j) * (DTRANK + 2 * DSTATE)
                       + DTRANK + c];
    }
    __syncthreads();
#pragma unroll 4
    for (int j = 0; j < TCHUNK; ++j) {
      size_t row = (size_t)(b * TT + t0 + j);
      float dtv = dt[row * DINNER + d];
      float uv  = uc[row * DINNER + d];
      float du  = dtv * uv;
      float y   = 0.0f;
#pragma unroll
      for (int s = 0; s < DSTATE; ++s) {
        h[s] = __expf(dtv * Ad[s]) * h[s] + du * bc[j][s];
        y += h[s] * bc[j][DSTATE + s];
      }
      ys[row * DINNER + d] = y;
    }
    __syncthreads();
  }
}

// y = (scan_y + uc*D) * silu(z)  -> bf16 for out_proj
__global__ void combine_kernel(const float* __restrict__ ys,
                               const float* __restrict__ uc,
                               const float* __restrict__ Dd,   // (DINNER)
                               const float* __restrict__ xz,   // (ROWS, 2*DINNER)
                               __bf16* __restrict__ ybf) {
  int idx = blockIdx.x * blockDim.x + threadIdx.x;
  if (idx >= ROWS * DINNER) return;
  int d   = idx % DINNER;
  int row = idx / DINNER;
  float y = ys[idx] + uc[idx] * Dd[d];
  float z = xz[(size_t)row * (2 * DINNER) + DINNER + d];
  y *= z * sigmoidf_(z);
  ybf[idx] = f2bf(y);
}

// final layernorm over DMODEL per row
__global__ void layernorm_kernel(const float* __restrict__ x,
                                 const float* __restrict__ w,
                                 const float* __restrict__ bias,
                                 float* __restrict__ out) {
  __shared__ float s1[256], s2[256];
  const int row = blockIdx.x;
  float sum = 0.0f, sq = 0.0f;
  for (int c = threadIdx.x; c < DMODEL; c += blockDim.x) {
    float v = x[(size_t)row * DMODEL + c];
    sum += v; sq += v * v;
  }
  s1[threadIdx.x] = sum; s2[threadIdx.x] = sq;
  __syncthreads();
  for (int off = 128; off > 0; off >>= 1) {
    if (threadIdx.x < (unsigned)off) {
      s1[threadIdx.x] += s1[threadIdx.x + off];
      s2[threadIdx.x] += s2[threadIdx.x + off];
    }
    __syncthreads();
  }
  float mean = s1[0] / DMODEL;
  float var  = s2[0] / DMODEL - mean * mean;
  float rstd = rsqrtf(var + 1e-5f);
  for (int c = threadIdx.x; c < DMODEL; c += blockDim.x) {
    float v = x[(size_t)row * DMODEL + c];
    out[(size_t)row * DMODEL + c] = (v - mean) * rstd * w[c] + bias[c];
  }
}

// ---------------------------------------------------------------------------
// Host orchestration
// ---------------------------------------------------------------------------
extern "C" void kernel_launch(void* const* d_in, const int* in_sizes, int n_in,
                              void* d_out, int out_size, void* d_ws, size_t ws_size,
                              hipStream_t stream) {
  (void)in_sizes; (void)n_in; (void)out_size; (void)ws_size;

  const float* x_in     = (const float*)d_in[0];
  const float* in_w     = (const float*)d_in[1];
  const float* conv_w   = (const float*)d_in[2];
  const float* conv_b   = (const float*)d_in[3];
  const float* xp_w     = (const float*)d_in[4];
  const float* dt_w     = (const float*)d_in[5];
  const float* dt_b     = (const float*)d_in[6];
  const float* A_log    = (const float*)d_in[7];
  const float* Dd       = (const float*)d_in[8];
  const float* out_w    = (const float*)d_in[9];
  const float* norm_w   = (const float*)d_in[10];
  const float* norm_b   = (const float*)d_in[11];
  float* out = (float*)d_out;

  // bump allocator over d_ws
  char* ws = (char*)d_ws;
  size_t off = 0;
  auto alloc = [&](size_t bytes) -> void* {
    void* p = ws + off;
    off = (off + bytes + 255) & ~(size_t)255;
    return p;
  };

  const int n_in_w  = NLAYERS * 2 * DINNER * DMODEL;   // 4*4096*1024
  const int n_xp_w  = NLAYERS * (DTRANK + 2 * DSTATE) * DINNER;
  const int n_dt_w  = NLAYERS * DINNER * DTRANK;
  const int n_out_w = NLAYERS * DMODEL * DINNER;

  __bf16* in_w_bf  = (__bf16*)alloc((size_t)n_in_w  * 2);
  __bf16* xp_w_bf  = (__bf16*)alloc((size_t)n_xp_w  * 2);
  __bf16* dt_w_bf  = (__bf16*)alloc((size_t)n_dt_w  * 2);
  __bf16* out_w_bf = (__bf16*)alloc((size_t)n_out_w * 2);

  float*  xcur  = (float*) alloc((size_t)ROWS * DMODEL * 4);
  __bf16* xbf   = (__bf16*)alloc((size_t)ROWS * DMODEL * 2);
  float*  xz    = (float*) alloc((size_t)ROWS * 2 * DINNER * 4);
  float*  uc    = (float*) alloc((size_t)ROWS * DINNER * 4);
  __bf16* ucbf  = (__bf16*)alloc((size_t)ROWS * DINNER * 2);
  float*  xdbl  = (float*) alloc((size_t)ROWS * (DTRANK + 2 * DSTATE) * 4);
  __bf16* dtrbf = (__bf16*)alloc((size_t)ROWS * DTRANK * 2);
  float*  dtbuf = (float*) alloc((size_t)ROWS * DINNER * 4);
  float*  ysbuf = (float*) alloc((size_t)ROWS * DINNER * 4);
  __bf16* ybf   = (__bf16*)alloc((size_t)ROWS * DINNER * 2);

  const int TPB = 256;
  auto nb = [](int n, int tpb) { return (n + tpb - 1) / tpb; };

  // weights -> bf16 (once per call; deterministic)
  cvt_f32_bf16_kernel<<<nb(n_in_w,  TPB), TPB, 0, stream>>>(in_w,  in_w_bf,  n_in_w);
  cvt_f32_bf16_kernel<<<nb(n_xp_w,  TPB), TPB, 0, stream>>>(xp_w,  xp_w_bf,  n_xp_w);
  cvt_f32_bf16_kernel<<<nb(n_dt_w,  TPB), TPB, 0, stream>>>(dt_w,  dt_w_bf,  n_dt_w);
  cvt_f32_bf16_kernel<<<nb(n_out_w, TPB), TPB, 0, stream>>>(out_w, out_w_bf, n_out_w);

  // residual stream starts as the input
  hipMemcpyAsync(xcur, x_in, (size_t)ROWS * DMODEL * 4,
                 hipMemcpyDeviceToDevice, stream);

  for (int l = 0; l < NLAYERS; ++l) {
    const __bf16* in_w_l  = in_w_bf  + (size_t)l * 2 * DINNER * DMODEL;
    const float*  cw_l    = conv_w   + (size_t)l * DINNER * DCONV;
    const float*  cb_l    = conv_b   + (size_t)l * DINNER;
    const __bf16* xp_w_l  = xp_w_bf  + (size_t)l * (DTRANK + 2 * DSTATE) * DINNER;
    const __bf16* dt_w_l  = dt_w_bf  + (size_t)l * DINNER * DTRANK;
    const float*  dtb_l   = dt_b     + (size_t)l * DINNER;
    const float*  Alog_l  = A_log    + (size_t)l * DINNER * DSTATE;
    const float*  Dd_l    = Dd       + (size_t)l * DINNER;
    const __bf16* out_w_l = out_w_bf + (size_t)l * DMODEL * DINNER;

    // x -> bf16 for in_proj A operand
    cvt_f32_bf16_kernel<<<nb(ROWS * DMODEL, TPB), TPB, 0, stream>>>(
        xcur, xbf, ROWS * DMODEL);

    // in_proj: (2048 x 1024) * (4096 x 1024)^T -> xz (2048 x 4096)
    {
      dim3 grid(2 * DINNER / 64, ROWS / 256);   // 64x8
      wmma_gemm_bf16<2, 4, 2 * DINNER, DMODEL, false><<<grid, 256, 0, stream>>>(
          xbf, in_w_l, xz);
    }

    // depthwise conv + bias + silu
    conv_silu_kernel<<<nb(ROWS * DINNER, TPB), TPB, 0, stream>>>(
        xz, cw_l, cb_l, uc, ucbf);

    // x_proj: (2048 x 2048) * (96 x 2048)^T -> x_dbl (2048 x 96)
    {
      dim3 grid((DTRANK + 2 * DSTATE) / 32, ROWS / 256);  // 3x8
      wmma_gemm_bf16<2, 2, DTRANK + 2 * DSTATE, DINNER, false>
          <<<grid, 256, 0, stream>>>(ucbf, xp_w_l, xdbl);
    }

    // dt_r slice -> bf16
    dtr_extract_kernel<<<nb(ROWS * DTRANK, TPB), TPB, 0, stream>>>(xdbl, dtrbf);

    // dt_proj: (2048 x 64) * (2048 x 64)^T -> dt_lin (2048 x 2048)
    {
      dim3 grid(DINNER / 64, ROWS / 256);       // 32x8
      wmma_gemm_bf16<2, 4, DINNER, DTRANK, false><<<grid, 256, 0, stream>>>(
          dtrbf, dt_w_l, dtbuf);
    }

    // dt = softplus(dt_lin + bias)
    softplus_kernel<<<nb(ROWS * DINNER, TPB), TPB, 0, stream>>>(dtbuf, dtb_l);

    // selective scan
    {
      dim3 grid(DINNER / TPB, BB);
      scan_kernel<<<grid, TPB, 0, stream>>>(xdbl, dtbuf, uc, Alog_l, ysbuf);
    }

    // y = (scan_y + uc*D) * silu(z)
    combine_kernel<<<nb(ROWS * DINNER, TPB), TPB, 0, stream>>>(
        ysbuf, uc, Dd_l, xz, ybf);

    // out_proj with fused residual: xcur += (2048x2048)*(1024x2048)^T
    {
      dim3 grid(DMODEL / 64, ROWS / 256);       // 16x8
      wmma_gemm_bf16<2, 4, DMODEL, DINNER, true><<<grid, 256, 0, stream>>>(
          ybf, out_w_l, xcur);
    }
  }

  // final layernorm -> d_out
  layernorm_kernel<<<ROWS, 256, 0, stream>>>(xcur, norm_w, norm_b, out);
}
